// CTCThenCIF_33449205301771
// MI455X (gfx1250) — matrix operand order repaired and checked by reading, not compiled, not run on hardware
//
#include <hip/hip_runtime.h>
#include <hip/hip_bf16.h>

// ---------------- problem constants ----------------
constexpr int Bq    = 16;
constexpr int Tq    = 4096;
constexpr int Dq    = 512;
constexpr int Vq    = 512;
constexpr int Hq    = 128;
constexpr int NMAX  = 400;

typedef __attribute__((ext_vector_type(16))) __bf16 bf16x16;
typedef __attribute__((ext_vector_type(8)))  __bf16 bf16x8;
typedef __attribute__((ext_vector_type(8)))  float  f32x8;

// =====================================================================
// K0: convert W1 [D,H] and Wv [D,V] (row-major f32) to column-major bf16
//     Wcm[col*D + k]  -> each WMMA B-fragment lane reads 16 contiguous K.
// =====================================================================
__global__ void cif_conv_weights(const float* __restrict__ W1,
                                 const float* __restrict__ Wv,
                                 __bf16* __restrict__ W1cm,
                                 __bf16* __restrict__ Wvcm) {
    int idx = blockIdx.x * 256 + threadIdx.x;
    if (idx < Vq * Dq) {
        int col = idx / Dq, k = idx % Dq;
        Wvcm[idx] = (__bf16)Wv[(size_t)k * Vq + col];
    } else {
        int j = idx - Vq * Dq;
        if (j < Hq * Dq) {
            int col = j / Dq, k = j % Dq;
            W1cm[j] = (__bf16)W1[(size_t)k * Hq + col];
        }
    }
}

// =====================================================================
// K1: fused per-16-row block:
//   LayerNorm stats -> bf16 A-tiles (raw + normalized) in LDS (row-major)
//   GEMM1: v = h @ Wv + bv        (8 waves x 4 tiles x 16 K-steps WMMA,
//                                  B fragments double-buffered in VGPRs)
//   GEMM2: x = LN(h) @ W1 + b1    (8 waves x 1 tile  x 16 K-steps WMMA)
//   SiLU(x) . W2 + b2 -> softplus -> raw_alpha
// =====================================================================
__global__ void __launch_bounds__(256)
cif_fused_gemm(const float* __restrict__ h,
               const float* __restrict__ ln_g, const float* __restrict__ ln_b,
               const float* __restrict__ b1,   const float* __restrict__ W2,
               const float* __restrict__ b2,   const float* __restrict__ bv,
               const __bf16* __restrict__ Wvcm, const __bf16* __restrict__ W1cm,
               float* __restrict__ vmat, float* __restrict__ rawOut) {
    __shared__ __bf16 sAraw[16][Dq];
    __shared__ __bf16 sAln [16][Dq];
    __shared__ float  sRed1[16][16], sRed2[16][16];
    __shared__ float  sMu[16], sRstd[16];
    __shared__ float  sXp[16][Hq];

    const int tid  = threadIdx.x;
    const int row  = tid >> 4;          // 0..15 : matrix row within block
    const int lg   = tid & 15;          // 0..15 : column group
    const size_t row0 = (size_t)blockIdx.x * 16;

    // ---- pass A: load 32 strided floats / thread, stats + raw bf16 ----
    const float* hrow = h + (row0 + row) * Dq;
    float xv[32];
    float s1 = 0.f, s2 = 0.f;
#pragma unroll
    for (int i = 0; i < 32; ++i) {
        int c = lg + 16 * i;
        float v = hrow[c];
        xv[i] = v; s1 += v; s2 += v * v;
        sAraw[row][c] = (__bf16)v;
    }
    sRed1[row][lg] = s1; sRed2[row][lg] = s2;
    __syncthreads();
    if (tid < 16) {
        float a = 0.f, q = 0.f;
#pragma unroll
        for (int j = 0; j < 16; ++j) { a += sRed1[tid][j]; q += sRed2[tid][j]; }
        float mu  = a * (1.0f / Dq);
        float var = q * (1.0f / Dq) - mu * mu;
        sMu[tid] = mu; sRstd[tid] = rsqrtf(var + 1e-5f);
    }
    __syncthreads();

    // ---- pass B: normalized bf16 tile ----
    {
        float mu = sMu[row], rs = sRstd[row];
#pragma unroll
        for (int i = 0; i < 32; ++i) {
            int c = lg + 16 * i;
            float nv = (xv[i] - mu) * rs * ln_g[c] + ln_b[c];
            sAln[row][c] = (__bf16)nv;
        }
    }
    __syncthreads();

    const int wv   = tid >> 5;   // wave 0..7
    const int lane = tid & 31;
    const int lrow = lane & 15;  // A row / B col / C col
    const int lhi  = lane >> 4;  // half-wave select

    // ---- GEMM1: v tile 16x512, each wave does 4 16x16 tiles.
    //      B fragments for K-step kk+1 are prefetched into registers
    //      before the WMMAs for kk are issued (hides L2 latency). ----
    {
        const int n0 = wv * 64;
        // per-lane base: column (n0 + tI*16 + lrow), K offset lhi*16
        const __bf16* bbase = Wvcm + (size_t)(n0 + lrow) * Dq + lhi * 16;

        f32x8 acc[4];
#pragma unroll
        for (int tI = 0; tI < 4; ++tI) {
            float bvv = bv[n0 + tI * 16 + lrow];
#pragma unroll
            for (int r = 0; r < 8; ++r) acc[tI][r] = bvv;
        }

        bf16x16 bcur[4];
#pragma unroll
        for (int tI = 0; tI < 4; ++tI)
            bcur[tI] = *(const bf16x16*)(bbase + (size_t)tI * 16 * Dq);

#pragma unroll
        for (int kk = 0; kk < Dq / 32; ++kk) {
            // prefetch next K-step B fragments (wraps to 0 on last step)
            const int kn = (kk + 1) & (Dq / 32 - 1);
            bf16x16 bnxt[4];
#pragma unroll
            for (int tI = 0; tI < 4; ++tI)
                bnxt[tI] = *(const bf16x16*)(bbase + (size_t)tI * 16 * Dq + kn * 32);

            // A fragment from LDS (contiguous K: lane half h holds k = 16h..16h+15)
            const int k0 = kk * 32 + lhi * 16;
            bf16x8 alo = *(const bf16x8*)&sAraw[lrow][k0];
            bf16x8 ahi = *(const bf16x8*)&sAraw[lrow][k0 + 8];
            bf16x16 a = __builtin_shufflevector(alo, ahi,
                         0,1,2,3,4,5,6,7,8,9,10,11,12,13,14,15);

#pragma unroll
            for (int tI = 0; tI < 4; ++tI)
                acc[tI] = __builtin_amdgcn_wmma_f32_16x16x32_bf16(
                              false, a, false, bcur[tI], (short)0, acc[tI], false, false);
#pragma unroll
            for (int tI = 0; tI < 4; ++tI) bcur[tI] = bnxt[tI];
        }

        const int mB = lhi * 8;
#pragma unroll
        for (int tI = 0; tI < 4; ++tI) {
            int ncol = n0 + tI * 16 + lrow;
#pragma unroll
            for (int r = 0; r < 8; ++r)
                vmat[(row0 + mB + r) * Vq + ncol] = acc[tI][r];
        }
    }

    // ---- GEMM2: alpha hidden 16x128, each wave one 16x16 tile ----
    {
        const int n0 = wv * 16;
        const __bf16* bbase = W1cm + (size_t)(n0 + lrow) * Dq + lhi * 16;

        f32x8 acc;
        {
            float b1v = b1[n0 + lrow];
#pragma unroll
            for (int r = 0; r < 8; ++r) acc[r] = b1v;
        }

        bf16x16 bcur = *(const bf16x16*)bbase;
#pragma unroll
        for (int kk = 0; kk < Dq / 32; ++kk) {
            const int kn = (kk + 1) & (Dq / 32 - 1);
            bf16x16 bnxt = *(const bf16x16*)(bbase + kn * 32);

            const int k0 = kk * 32 + lhi * 16;
            bf16x8 alo = *(const bf16x8*)&sAln[lrow][k0];
            bf16x8 ahi = *(const bf16x8*)&sAln[lrow][k0 + 8];
            bf16x16 a = __builtin_shufflevector(alo, ahi,
                         0,1,2,3,4,5,6,7,8,9,10,11,12,13,14,15);

            acc = __builtin_amdgcn_wmma_f32_16x16x32_bf16(
                      false, a, false, bcur, (short)0, acc, false, false);
            bcur = bnxt;
        }
        float w2v = W2[n0 + lrow];
        const int mB = lhi * 8;
#pragma unroll
        for (int r = 0; r < 8; ++r) {
            float x = acc[r];
            float s = x / (1.0f + __expf(-x));         // SiLU
            sXp[mB + r][n0 + lrow] = s * w2v;
        }
    }
    __syncthreads();

    if (tid < 16) {
        float s = 0.f;
#pragma unroll 8
        for (int j = 0; j < Hq; ++j) s += sXp[tid][j];
        s += b2[0];
        float sp = (s > 20.f) ? s : ((s < -20.f) ? __expf(s) : log1pf(__expf(s)));
        rawOut[row0 + tid] = sp;
    }
}

// =====================================================================
// K2: per-batch mask + rescale + inclusive cumsum (one block per batch)
// =====================================================================
__global__ void __launch_bounds__(256)
cif_alpha_scan(const float* __restrict__ raw,
               const int* __restrict__ tgt, const int* __restrict__ inlen,
               float* __restrict__ alphaOut, float* __restrict__ cOut) {
    __shared__ float ssum[256];
    const int b = blockIdx.x, tid = threadIdx.x;
    const int L = inlen[b];
    const int base = tid * 16;
    float m[16];
    float tsum = 0.f;
#pragma unroll
    for (int i = 0; i < 16; ++i) {
        int t = base + i;
        float v = raw[(size_t)b * Tq + t];
        v = (t < L) ? v : 0.f;
        m[i] = v; tsum += v;
    }
    ssum[tid] = tsum;
    __syncthreads();
    for (int d = 1; d < 256; d <<= 1) {
        float v = (tid >= d) ? ssum[tid - d] : 0.f;
        __syncthreads();
        ssum[tid] += v;
        __syncthreads();
    }
    const float excl  = ssum[tid] - tsum;
    const float total = ssum[255];

    const float Nf      = (float)tgt[b];
    const float scale   = Nf / (total + 1e-8f);
    const bool  uniform = (total < 1e-7f);          // 10 * EPS
    const float Lf      = fmaxf((float)L, 1.f);
    const float unif    = Nf / Lf;

    float run = excl;
#pragma unroll
    for (int i = 0; i < 16; ++i) {
        int t = base + i;
        float a, c;
        if (!uniform) {
            a = m[i] * scale;
            run += m[i];
            c = run * scale;
        } else {
            float msk = (t < L) ? 1.f : 0.f;
            a = msk * unif;
            int cnt = (t + 1 < L) ? (t + 1) : L;
            c = (float)cnt * unif;
        }
        alphaOut[(size_t)b * Tq + t] = a;
        cOut[(size_t)b * Tq + t]     = c;
    }
}

// =====================================================================
// K3: sparse mass-overlap scatter; one block per (token bin n, batch b)
//     frames overlapping bin n form a contiguous range on monotone c.
// =====================================================================
__global__ void __launch_bounds__(256)
cif_scatter(const float* __restrict__ vmat, const float* __restrict__ alphaA,
            const float* __restrict__ cArr, const int* __restrict__ tgt,
            float* __restrict__ tokens, float* __restrict__ durs,
            float* __restrict__ mass,   float* __restrict__ maskout) {
    const int n = blockIdx.x, b = blockIdx.y, tid = threadIdx.x;
    const float kf = (float)(n + 1);
    const float* cb = cArr + (size_t)b * Tq;
    const float* ab = alphaA + (size_t)b * Tq;

    __shared__ int sLo, sHi;
    if (tid == 0) {
        int l = 0, r = Tq;                 // first t with c[t] > k-1
        while (l < r) { int mid = (l + r) >> 1; if (cb[mid] > kf - 1.0f) r = mid; else l = mid + 1; }
        sLo = l;
        l = 0; r = Tq;                     // first s with c[s] >= k  -> hi = s+1
        while (l < r) { int mid = (l + r) >> 1; if (cb[mid] >= kf) r = mid; else l = mid + 1; }
        int hi = l + 1; if (hi > Tq) hi = Tq;
        sHi = hi;
    }
    __syncthreads();
    const int lo = sLo, hi = sHi;

    float acc0 = 0.f, acc1 = 0.f, wm = 0.f, wd = 0.f;
    for (int t = lo; t < hi; ++t) {
        float ct = cb[t];
        float at = ab[t];
        float cp = ct - at;
        float w = fminf(ct, kf) - fmaxf(cp, kf - 1.0f);
        w = fmaxf(w, 0.f);
        const float* vr = vmat + ((size_t)b * Tq + t) * Vq;
        acc0 += w * vr[tid];
        acc1 += w * vr[tid + 256];
        wm += w;
        wd += w / (at + 1e-8f);
    }
    const float tm = (n < tgt[b]) ? 1.f : 0.f;
    const size_t ob = ((size_t)b * NMAX + n) * Vq;
    tokens[ob + tid]       = tm * acc0;
    tokens[ob + tid + 256] = tm * acc1;
    if (tid == 0) {
        durs[b * NMAX + n]    = tm * wd;
        mass[b * NMAX + n]    = tm * wm;
        maskout[b * NMAX + n] = tm;
    }
}

// =====================================================================
extern "C" void kernel_launch(void* const* d_in, const int* in_sizes, int n_in,
                              void* d_out, int out_size, void* d_ws, size_t ws_size,
                              hipStream_t stream) {
    const float* h     = (const float*)d_in[0];
    const int*   tgt   = (const int*)  d_in[1];
    const int*   inlen = (const int*)  d_in[2];
    const float* ln_g  = (const float*)d_in[3];
    const float* ln_b  = (const float*)d_in[4];
    const float* W1    = (const float*)d_in[5];
    const float* b1    = (const float*)d_in[6];
    const float* W2    = (const float*)d_in[7];
    const float* b2    = (const float*)d_in[8];
    const float* Wv    = (const float*)d_in[9];
    const float* bv    = (const float*)d_in[10];

    // outputs, concatenated flat in reference return order
    float* out     = (float*)d_out;
    float* tokens  = out;                                   // [16,400,512]
    float* durs    = tokens + (size_t)Bq * NMAX * Vq;       // [16,400]
    float* massO   = durs + (size_t)Bq * NMAX;              // [16,400]
    float* tmaskO  = massO + (size_t)Bq * NMAX;             // [16,400]
    float* alphaO  = tmaskO + (size_t)Bq * NMAX;            // [16,4096]
    float* rawO    = alphaO + (size_t)Bq * Tq;              // [16,4096]

    // workspace layout
    char* w = (char*)d_ws;
    float*  vmat = (float*)w;                                          // 134 MB
    size_t  off  = (size_t)Bq * Tq * Vq * sizeof(float);
    __bf16* Wvcm = (__bf16*)(w + off);  off += (size_t)Vq * Dq * sizeof(__bf16);
    __bf16* W1cm = (__bf16*)(w + off);  off += (size_t)Hq * Dq * sizeof(__bf16);
    float*  cbuf = (float*)(w + off);                                   // [16,4096]

    cif_conv_weights<<<(Vq * Dq + Hq * Dq + 255) / 256, 256, 0, stream>>>(W1, Wv, W1cm, Wvcm);
    cif_fused_gemm<<<(Bq * Tq) / 16, 256, 0, stream>>>(h, ln_g, ln_b, b1, W2, b2, bv,
                                                       Wvcm, W1cm, vmat, rawO);
    cif_alpha_scan<<<Bq, 256, 0, stream>>>(rawO, tgt, inlen, alphaO, cbuf);
    cif_scatter<<<dim3(NMAX, Bq), 256, 0, stream>>>(vmat, alphaO, cbuf, tgt,
                                                    tokens, durs, massO, tmaskO);
}
